// PointNet_14525579395243
// MI455X (gfx1250) — matrix operand order repaired and checked by reading, not compiled, not run on hardware
//
#include <hip/hip_runtime.h>

typedef __attribute__((ext_vector_type(16))) _Float16 v16h;
typedef __attribute__((ext_vector_type(8)))  _Float16 v8h;
typedef __attribute__((ext_vector_type(8)))  float    v8f;

#define NPTS   1000000
#define NTILES (NPTS / 16)
#define NGRAPH 16
#define GCW    256   // pooled feature width (32*base)

// ---------------------------------------------------------------- WMMA core
__device__ __forceinline__ v8f wmma_f16(v16h a, v16h b, v8f c) {
  // v_wmma_f32_16x16x32_f16  (8-arg form: neg_a, A, neg_b, B, c_mod, C, reuse_a, reuse_b)
  return __builtin_amdgcn_wmma_f32_16x16x32_f16(false, a, false, b, (short)0, c, false, false);
}

union V16U { v16h v; v8h h[2]; };

// A fragment (16x32 f16, M x K): lane L holds row M=L&15.
// lanes 0-15: K = k0+0..7 (v0-3) and k0+16..23 (v4-7); lanes 16-31: +8 / +24.
template<int STRIDE>
__device__ __forceinline__ v16h loadA(const _Float16* act, int c, int lane) {
  const int r = lane & 15, half = lane >> 4;
  const _Float16* p = act + r * STRIDE + c * 32 + half * 8;
  V16U u;
  u.h[0] = *(const v8h*)p;         // ds_load_b128
  u.h[1] = *(const v8h*)(p + 16);  // ds_load_b128
  return u.v;
}

// B fragment (32x16 f16, K x N) from transposed weights wT[N][KPAD]:
// lane L holds column N = n0 + (L&15); lanes 0-15: K = k0+0..7 and k0+16..23; lanes 16-31: +8 / +24.
template<int KPAD>
__device__ __forceinline__ v16h loadB(const _Float16* wT, int nt, int c, int lane) {
  const int r = lane & 15, half = lane >> 4;
  const _Float16* p = wT + (nt * 16 + r) * KPAD + c * 32 + half * 8;
  V16U u;
  u.h[0] = *(const v8h*)p;
  u.h[1] = *(const v8h*)(p + 16);
  return u.v;
}

// One dense layer on a 16-row tile living in per-wave LDS (row-major, STRIDE f16 pitch).
// Preloads all A chunks to registers, so in-place output write is safe.
template<int STRIDE, int KPAD, int NOUT>
__device__ __forceinline__ void mlp_layer(const _Float16* __restrict__ wT,
                                          const float* __restrict__ bias,
                                          _Float16* act, int lane, bool relu) {
  constexpr int NC = KPAD / 32;
  const int r = lane & 15, half = lane >> 4;
  asm volatile("" ::: "memory");
  v16h A[NC];
#pragma unroll
  for (int c = 0; c < NC; ++c) A[c] = loadA<STRIDE>(act, c, lane);
#pragma unroll
  for (int nt = 0; nt < NOUT / 16; ++nt) {
    v8f acc = {};
#pragma unroll
    for (int c = 0; c < NC; ++c) acc = wmma_f16(A[c], loadB<KPAD>(wT, nt, c, lane), acc);
    const float b = bias[nt * 16 + r];
#pragma unroll
    for (int v = 0; v < 8; ++v) {
      float y = acc[v] + b;
      if (relu) y = fmaxf(y, 0.0f);
      // D layout: lane owns col n, VGPR v owns row (v + 8*half)
      act[(half * 8 + v) * STRIDE + (nt * 16 + r)] = (_Float16)y;
    }
  }
}

// Zero a 16-column pad region of the activation tile (rows 0..15).
template<int STRIDE>
__device__ __forceinline__ void clear16(_Float16* act, int from, int lane) {
#pragma unroll
  for (int i = lane; i < 256; i += 32) {
    const int row = i >> 4, c = from + (i & 15);
    act[row * STRIDE + c] = (_Float16)0.0f;
  }
}

// Stage a [K][N] fp32 weight into LDS transposed+padded as f16 [NPAD][KPAD].
// Compile-time dims -> constant-divide (no VALU integer division).
template<int K, int N, int KPAD, int NPAD>
__device__ void stage_wT(_Float16* dst, const float* __restrict__ W, int tid, int nthr) {
  for (int idx = tid; idx < NPAD * KPAD; idx += nthr) {
    const int n = idx / KPAD, k = idx - n * KPAD;
    const float v = (k < K && n < N) ? W[k * N + n] : 0.0f;
    dst[idx] = (_Float16)v;
  }
}
template<int N, int NPAD>
__device__ void stage_bias(float* dst, const float* __restrict__ b, int tid, int nthr) {
  for (int i = tid; i < NPAD; i += nthr) dst[i] = (i < N) ? b[i] : 0.0f;
}

// Monotonic-bit-pattern float atomic max.
__device__ __forceinline__ void atomicMaxF(float* a, float v) {
  const int iv = __float_as_int(v);
  if (iv >= 0) atomicMax((int*)a, iv);
  else         atomicMin((unsigned int*)a, (unsigned int)iv);
}

// ---------------------------------------------------------------- kernels
__global__ void pointnet_init_gmax(float* g) {
  const int i = blockIdx.x * blockDim.x + threadIdx.x;
  if (i < NGRAPH * GCW) g[i] = __int_as_float(0xFF800000);  // -inf
}

struct P1W {
  const float *ew1, *eb1, *ew2, *eb2;
  const float *iw1, *ib1, *iw2, *ib2;
  const float *mw1, *mb1, *mw2, *mb2, *mw3, *mb3;
};

__global__ __launch_bounds__(256, 1)
void pointnet_pass1(const float* __restrict__ x, const int* __restrict__ batch, P1W W,
                    float* __restrict__ gmax, _Float16* __restrict__ zout) {
  extern __shared__ __align__(16) char smem[];
  _Float16* wE1 = (_Float16*)smem;       // 64x32
  _Float16* wE2 = wE1 + 64 * 32;         // 64x64
  _Float16* wI1 = wE2 + 64 * 64;         // 16x64
  _Float16* wI2 = wI1 + 16 * 64;         // 16x32
  _Float16* wM1 = wI2 + 16 * 32;         // 32x32
  _Float16* wM2 = wM1 + 32 * 32;         // 64x32
  _Float16* wM3 = wM2 + 64 * 32;         // 256x64
  float* bE1 = (float*)(wM3 + 256 * 64);
  float* bE2 = bE1 + 64;
  float* bI1 = bE2 + 64;
  float* bI2 = bI1 + 16;
  float* bM1 = bI2 + 16;
  float* bM2 = bM1 + 32;
  float* bM3 = bM2 + 64;
  _Float16* actBase = (_Float16*)(bM3 + 256);  // 8 waves * 16x64

  const int tid = threadIdx.x, nthr = blockDim.x;
  stage_wT<  7,  64, 32,  64>(wE1, W.ew1, tid, nthr);
  stage_wT< 64,  64, 64,  64>(wE2, W.ew2, tid, nthr);
  stage_wT< 64,   8, 64,  16>(wI1, W.iw1, tid, nthr);
  stage_wT<  8,  16, 32,  16>(wI2, W.iw2, tid, nthr);
  stage_wT< 16,  32, 32,  32>(wM1, W.mw1, tid, nthr);
  stage_wT< 32,  64, 32,  64>(wM2, W.mw2, tid, nthr);
  stage_wT< 64, 256, 64, 256>(wM3, W.mw3, tid, nthr);
  stage_bias< 64,  64>(bE1, W.eb1, tid, nthr);
  stage_bias< 64,  64>(bE2, W.eb2, tid, nthr);
  stage_bias<  8,  16>(bI1, W.ib1, tid, nthr);
  stage_bias< 16,  16>(bI2, W.ib2, tid, nthr);
  stage_bias< 32,  32>(bM1, W.mb1, tid, nthr);
  stage_bias< 64,  64>(bM2, W.mb2, tid, nthr);
  stage_bias<256, 256>(bM3, W.mb3, tid, nthr);
  __syncthreads();

  const int wave = tid >> 5, lane = tid & 31;
  const int r = lane & 15, half = lane >> 4;
  _Float16* act = actBase + wave * (16 * 64);
  const long long step = (long long)gridDim.x * 8;

  for (long long t = (long long)blockIdx.x * 8 + wave; t < NTILES; t += step) {
    const long long base = t * 16;
    __builtin_prefetch(x + (base + 16 * step) * 7, 0, 1);  // global_prefetch_b8 next tile

    // ---- encoder L1: K=7 (zero-padded to 32), N=64, relu
    v16h a0 = {};
    if (half == 0) {
      const float* xp = x + (base + r) * 7;
#pragma unroll
      for (int k = 0; k < 7; ++k) a0[k] = (_Float16)xp[k];
    }
#pragma unroll
    for (int nt = 0; nt < 4; ++nt) {
      v8f acc = {};
      acc = wmma_f16(a0, loadB<32>(wE1, nt, 0, lane), acc);
      const float b = bE1[nt * 16 + r];
#pragma unroll
      for (int v = 0; v < 8; ++v)
        act[(half * 8 + v) * 64 + nt * 16 + r] = (_Float16)fmaxf(acc[v] + b, 0.0f);
    }
    mlp_layer<64, 64, 64>(wE2, bE2, act, lane, false);  // encoder L2 (plain)
    mlp_layer<64, 64, 16>(wI1, bI1, act, lane, true);   // in L1 (relu), 8->pad16
    clear16<64>(act, 16, lane);
    mlp_layer<64, 32, 16>(wI2, bI2, act, lane, false);  // in L2 -> z (plain)

    {  // spill z (16 f16 cols) for pass 2
      const int row = lane >> 1, h2 = lane & 1;
      *(v8h*)(zout + base * 16 + row * 16 + h2 * 8) = *(const v8h*)(act + row * 64 + h2 * 8);
    }
    clear16<64>(act, 16, lane);
    mlp_layer<64, 32, 32>(wM1, bM1, act, lane, true);   // max L1
    mlp_layer<64, 32, 64>(wM2, bM2, act, lane, true);   // max L2

    // ---- max L3 (64 -> 256, plain) fused with per-graph segment-max
    const v16h A0 = loadA<64>(act, 0, lane);
    const v16h A1 = loadA<64>(act, 1, lane);
    const int  seg = batch[base + r];
    const bool uni = (__ballot(seg == __shfl(seg, 0, 32)) == 0xFFFFFFFFull);
#pragma unroll
    for (int nt = 0; nt < 16; ++nt) {
      v8f acc = {};
      acc = wmma_f16(A0, loadB<64>(wM3, nt, 0, lane), acc);
      acc = wmma_f16(A1, loadB<64>(wM3, nt, 1, lane), acc);
      const float b = bM3[nt * 16 + r];
      const int   c = nt * 16 + r;
      if (uni) {  // whole tile in one graph (common: batch is sorted)
        float m = acc[0] + b;
#pragma unroll
        for (int v = 1; v < 8; ++v) m = fmaxf(m, acc[v] + b);
        m = fmaxf(m, __shfl_xor(m, 16, 32));  // combine rows 0-7 with 8-15
        if (half == 0) atomicMaxF(&gmax[seg * GCW + c], m);
      } else {    // boundary tile: per-row atomics
#pragma unroll
        for (int v = 0; v < 8; ++v) {
          const int s = batch[base + half * 8 + v];
          atomicMaxF(&gmax[s * GCW + c], acc[v] + b);
        }
      }
    }
  }
}

struct P2W {
  const float *ow1, *ob1, *ow2, *ob2, *ow3, *ob3;
  const float *fw, *fb;
  const float *dw1, *db1, *dw2, *db2;
};

__global__ __launch_bounds__(128, 1)
void pointnet_pass2(const int* __restrict__ batch, P2W W,
                    const float* __restrict__ gmax, const _Float16* __restrict__ zin,
                    float* __restrict__ out) {
  extern __shared__ __align__(16) char smem[];
  _Float16* wO1 = (_Float16*)smem;        // 128x288
  _Float16* wO2 = wO1 + 128 * 288;        // 64x128
  _Float16* wO3 = wO2 + 64 * 128;         // 32x64
  _Float16* wF  = wO3 + 32 * 64;          // 64x32
  _Float16* wD1 = wF  + 64 * 32;          // 64x64
  _Float16* wD2 = wD1 + 64 * 64;          // 16x64
  float* bO1 = (float*)(wD2 + 16 * 64);   // 128
  float* bO2 = bO1 + 128;
  float* bO3 = bO2 + 64;
  float* bF  = bO3 + 32;
  float* bD1 = bF + 64;
  float* bD2 = bD1 + 64;                  // 16
  float*    gchF    = bD2 + 16;           // 16 x 256 pooled features (fp32, TDM target)
  _Float16* actBase = (_Float16*)(gchF + 16 * GCW);  // 4 waves * 16x288

  const int tid = threadIdx.x, nthr = blockDim.x;
  stage_wT<272, 128, 288, 128>(wO1, W.ow1, tid, nthr);
  stage_wT<128,  64, 128,  64>(wO2, W.ow2, tid, nthr);
  stage_wT< 64,  32,  64,  32>(wO3, W.ow3, tid, nthr);
  stage_wT< 32,  64,  32,  64>(wF,  W.fw,  tid, nthr);
  stage_wT< 64,  64,  64,  64>(wD1, W.dw1, tid, nthr);
  stage_wT< 64,   4,  64,  16>(wD2, W.dw2, tid, nthr);
  stage_bias<128, 128>(bO1, W.ob1, tid, nthr);
  stage_bias< 64,  64>(bO2, W.ob2, tid, nthr);
  stage_bias< 32,  32>(bO3, W.ob3, tid, nthr);
  stage_bias< 64,  64>(bF,  W.fb,  tid, nthr);
  stage_bias< 64,  64>(bD1, W.db1, tid, nthr);
  stage_bias<  4,  16>(bD2, W.db2, tid, nthr);

  // ---- pooled feature table gmax (16x256 fp32, 16 KB contiguous) -> LDS.
  // One Tensor-Data-Mover descriptor issued by wave 0; tracked by TENSORcnt.
#if defined(__gfx1250__) && __has_builtin(__builtin_amdgcn_tensor_load_to_lds) && \
    __has_builtin(__builtin_amdgcn_s_wait_tensorcnt)
  if (tid < 32) {  // uniform per-wave branch: only wave 0 issues the TDM op
    typedef unsigned int v4u __attribute__((ext_vector_type(4)));
    typedef int          v4i __attribute__((ext_vector_type(4)));
    typedef int          v8i __attribute__((ext_vector_type(8)));
    const unsigned long long ga  = (unsigned long long)(const void*)gmax;
    const unsigned           lds = (unsigned)(unsigned long long)(void*)gchF;
    // D# group0: count=1 | lds_addr | global_addr[56:0] | type=2
    v4u g0 = { 1u, lds,
               (unsigned)(ga & 0xffffffffull),
               (unsigned)((ga >> 32) & 0x01ffffffull) | (2u << 30) };
    // D# group1: data_size=4B; 1D tile: tensor_dim0 = tile_dim0 = 4096 elements
    v8i g1 = { (int)(2u << 16),      // workgroup_mask=0, data_size=2 (4 bytes)
               (int)(4096u << 16),   // tensor_dim0[15:0] @ bits 63:48
               (int)(1u << 16),      // tensor_dim1[15:0] = 1 @ bits 95:80
               (int)(4096u << 16),   // tile_dim0 = 4096 @ bits 127:112
               0,                    // tile_dim1 = tile_dim2 = 0 (unused)
               4096,                 // tensor_dim0_stride low 32
               (int)(4096u << 16),   // tensor_dim1_stride low 16 @ bits 223:208
               0 };
    v4i z4 = {0, 0, 0, 0};
#if __clang_major__ >= 23
    v8i z8 = {0, 0, 0, 0, 0, 0, 0, 0};
    __builtin_amdgcn_tensor_load_to_lds(g0, g1, z4, z4, z8, 0);
#else
    __builtin_amdgcn_tensor_load_to_lds(g0, g1, z4, z4, 0);
#endif
    __builtin_amdgcn_s_wait_tensorcnt(0);  // s_wait_tensorcnt 0x0
  }
#else
  for (int i = tid; i < 16 * GCW; i += nthr) gchF[i] = gmax[i];
#endif
  __syncthreads();

  const int wave = tid >> 5, lane = tid & 31;
  const int r = lane & 15, half = lane >> 4;
  _Float16* act = actBase + wave * (16 * 288);
  const long long step = (long long)gridDim.x * 4;

  for (long long t = (long long)blockIdx.x * 4 + wave; t < NTILES; t += step) {
    const long long base = t * 16;
    {  // cols 0..15 <- z
      const int row = lane >> 1, h2 = lane & 1;
      *(v8h*)(act + row * 288 + h2 * 8) = *(const v8h*)(zin + base * 16 + row * 16 + h2 * 8);
    }
#pragma unroll 4
    for (int row = 0; row < 16; ++row) {  // cols 16..271 <- gmax[batch[i]] (f32 -> f16)
      const int s = batch[base + row];    // scalar (uniform across lanes)
      const float* src = gchF + s * GCW + lane * 8;
      v8h o;
#pragma unroll
      for (int j = 0; j < 8; ++j) o[j] = (_Float16)src[j];
      *(v8h*)(act + row * 288 + 16 + lane * 8) = o;
    }
    clear16<288>(act, 272, lane);         // pad 272..287

    mlp_layer<288, 288, 128>(wO1, bO1, act, lane, true);   // out L1
    mlp_layer<288, 128,  64>(wO2, bO2, act, lane, true);   // out L2
    mlp_layer<288,  64,  32>(wO3, bO3, act, lane, false);  // out L3 (plain)
    mlp_layer<288,  32,  64>(wF,  bF,  act, lane, false);  // fcfinal (plain)
    mlp_layer<288,  64,  64>(wD1, bD1, act, lane, true);   // decoder L1

    // decoder L2: 64 -> 4 (padded N=16), store straight to output (fp32)
    const v16h A0 = loadA<288>(act, 0, lane);
    const v16h A1 = loadA<288>(act, 1, lane);
    v8f acc = {};
    acc = wmma_f16(A0, loadB<64>(wD2, 0, 0, lane), acc);
    acc = wmma_f16(A1, loadB<64>(wD2, 0, 1, lane), acc);
    if (r < 4) {
      const float b = bD2[r];
#pragma unroll
      for (int v = 0; v < 8; ++v)
        out[(base + half * 8 + v) * 4 + r] = acc[v] + b;
    }
  }
}

// ---------------------------------------------------------------- launch
extern "C" void kernel_launch(void* const* d_in, const int* in_sizes, int n_in,
                              void* d_out, int out_size, void* d_ws, size_t ws_size,
                              hipStream_t stream) {
  (void)in_sizes; (void)n_in; (void)out_size; (void)ws_size;
  const float* x     = (const float*)d_in[0];
  const int*   batch = (const int*)d_in[1];

  P1W w1 = { (const float*)d_in[2],  (const float*)d_in[3],  (const float*)d_in[4],  (const float*)d_in[5],
             (const float*)d_in[6],  (const float*)d_in[7],  (const float*)d_in[8],  (const float*)d_in[9],
             (const float*)d_in[10], (const float*)d_in[11], (const float*)d_in[12], (const float*)d_in[13],
             (const float*)d_in[14], (const float*)d_in[15] };
  P2W w2 = { (const float*)d_in[16], (const float*)d_in[17], (const float*)d_in[18], (const float*)d_in[19],
             (const float*)d_in[20], (const float*)d_in[21], (const float*)d_in[22], (const float*)d_in[23],
             (const float*)d_in[24], (const float*)d_in[25], (const float*)d_in[26], (const float*)d_in[27] };

  float*     gmax = (float*)d_ws;                                            // 16x256 fp32
  _Float16*  zbuf = (_Float16*)((char*)d_ws + NGRAPH * GCW * sizeof(float)); // N x 16 f16
  float*     outp = (float*)d_out;

  constexpr size_t SM1 =
      (64*32 + 64*64 + 16*64 + 16*32 + 32*32 + 64*32 + 256*64) * 2 +
      (64 + 64 + 16 + 16 + 32 + 64 + 256) * 4 + 8 * 16 * 64 * 2;             // ~71 KB
  constexpr size_t SM2 =
      (128*288 + 64*128 + 32*64 + 64*32 + 64*64 + 16*64) * 2 +
      (128 + 64 + 32 + 64 + 64 + 16) * 4 + 16 * GCW * 4 + 4 * 16 * 288 * 2;  // ~159.4 KB

  hipFuncSetAttribute((const void*)pointnet_pass1,
                      hipFuncAttributeMaxDynamicSharedMemorySize, (int)SM1);
  hipFuncSetAttribute((const void*)pointnet_pass2,
                      hipFuncAttributeMaxDynamicSharedMemorySize, (int)SM2);

  pointnet_init_gmax<<<dim3((NGRAPH * GCW + 255) / 256), dim3(256), 0, stream>>>(gmax);
  pointnet_pass1<<<dim3(1280), dim3(256), SM1, stream>>>(x, batch, w1, gmax, zbuf);
  pointnet_pass2<<<dim3(1280), dim3(128), SM2, stream>>>(batch, w2, gmax, zbuf, outp);
}
